// MultiHeadedAttention_6098853560550
// MI455X (gfx1250) — matrix operand order repaired and checked by reading, not compiled, not run on hardware
//
#include <hip/hip_runtime.h>

typedef __attribute__((ext_vector_type(16))) _Float16 v16h;
typedef __attribute__((ext_vector_type(8)))  float    v8f;
typedef __attribute__((ext_vector_type(4)))  int      i4;

#define B_  4
#define S_  2048
#define D_  1024
#define H_  16
#define DK_ 64
#define NEGV (-1000000000.0f)

#if defined(__gfx1250__) && \
    __has_builtin(__builtin_amdgcn_global_load_async_to_lds_b128) && \
    __has_builtin(__builtin_amdgcn_s_wait_asynccnt)
#define USE_ASYNC 1
#else
#define USE_ASYNC 0
#endif

#if USE_ASYNC
// One 16-byte granule, DMA'd global -> LDS, tracked by ASYNCcnt (no VGPR staging).
__device__ __forceinline__ void async_cp16(void* lds_dst, const void* gsrc) {
  __builtin_amdgcn_global_load_async_to_lds_b128(
      (i4*)gsrc,
      (__attribute__((address_space(3))) i4*)lds_dst,
      0, 0);
}
#define ASYNC_WAIT() __builtin_amdgcn_s_wait_asynccnt(0)
#else
#define ASYNC_WAIT()
#endif

// ---------- WMMA fragment loaders (contiguous k order: k = 16*(lane/16) + s, wave32) ----------
__device__ __forceinline__ v16h load_A_frag(const _Float16* base, int lda) {
  const int lane = threadIdx.x & 31;
  const _Float16* p = base + (lane & 15) * lda + ((lane >> 4) << 4);
  v16h a;
#pragma unroll
  for (int i = 0; i < 8; ++i) { a[i] = p[i]; a[8 + i] = p[8 + i]; }
  return a;
}
__device__ __forceinline__ v16h load_B_frag(const _Float16* base, int ldb) {
  const int lane = threadIdx.x & 31;
  const _Float16* p = base + (lane & 15) * ldb + ((lane >> 4) << 4);
  v16h b;
#pragma unroll
  for (int i = 0; i < 16; ++i) b[i] = p[i];
  return b;
}

__device__ __forceinline__ float red16_max(float x) {
#pragma unroll
  for (int m = 1; m < 16; m <<= 1) x = fmaxf(x, __shfl_xor(x, m, 32));
  return x;
}
__device__ __forceinline__ float red16_sum(float x) {
#pragma unroll
  for (int m = 1; m < 16; m <<= 1) x += __shfl_xor(x, m, 32);
  return x;
}

// ---------------- Kernel 1: fused QKV projections (f32 in -> f16 out) ----------------
__global__ __launch_bounds__(256)
void proj_gemm_f32(const float* __restrict__ Xq, const float* __restrict__ Xk,
                   const float* __restrict__ Xv,
                   const float* __restrict__ Wq, const float* __restrict__ Wk,
                   const float* __restrict__ Wv,
                   const float* __restrict__ bq, const float* __restrict__ bk,
                   const float* __restrict__ bv,
                   _Float16* __restrict__ Yq, _Float16* __restrict__ Yk,
                   _Float16* __restrict__ Yv) {
  __shared__ _Float16 As[128 * 40];   // [m][k] k-contig
  __shared__ _Float16 Bs[128 * 40];   // [n][k] k-contig (W transposed)
  const int tid  = threadIdx.x;
  const int lane = tid & 31;
  const int wid  = tid >> 5;
  const int wm = (wid >> 2) * 64;
  const int wn = (wid & 3) * 32;
  const int bn = blockIdx.x * 128;
  const int bm = blockIdx.y * 128;

  const float* X; const float* W; const float* bias; _Float16* Y;
  if (blockIdx.z == 0)      { X = Xq; W = Wq; bias = bq; Y = Yq; }
  else if (blockIdx.z == 1) { X = Xk; W = Wk; bias = bk; Y = Yk; }
  else                      { X = Xv; W = Wv; bias = bv; Y = Yv; }

  v8f vzero = {};
  v8f acc[4][2];
#pragma unroll
  for (int i = 0; i < 4; ++i)
#pragma unroll
    for (int j = 0; j < 2; ++j) acc[i][j] = vzero;

  for (int k0 = 0; k0 < D_; k0 += 32) {
    __syncthreads();
#pragma unroll
    for (int i = 0; i < 16; ++i) {          // A tile 128x32, f32->f16 (conversion: manual)
      int l = tid + i * 256;
      int r = l >> 5, c = l & 31;
      As[r * 40 + c] = (_Float16)X[(bm + r) * D_ + (k0 + c)];
    }
#pragma unroll
    for (int i = 0; i < 16; ++i) {          // W tile 32x128 -> transpose (conversion: manual)
      int l = tid + i * 256;
      int n = l & 127, kk = l >> 7;
      Bs[n * 40 + kk] = (_Float16)W[(k0 + kk) * D_ + (bn + n)];
    }
    __syncthreads();
    v16h af[4], bf[2];
#pragma unroll
    for (int i = 0; i < 4; ++i) af[i] = load_A_frag(&As[(wm + i * 16) * 40], 40);
#pragma unroll
    for (int j = 0; j < 2; ++j) bf[j] = load_B_frag(&Bs[(wn + j * 16) * 40], 40);
#pragma unroll
    for (int i = 0; i < 4; ++i)
#pragma unroll
      for (int j = 0; j < 2; ++j)
        acc[i][j] = __builtin_amdgcn_wmma_f32_16x16x32_f16(
            false, af[i], false, bf[j], (short)0, acc[i][j], false, false);
  }

  const int g = lane >> 4, nlo = lane & 15;
#pragma unroll
  for (int i = 0; i < 4; ++i)
#pragma unroll
    for (int j = 0; j < 2; ++j) {
      int col = bn + wn + j * 16 + nlo;
      float bval = bias[col];
#pragma unroll
      for (int v = 0; v < 8; ++v) {
        int row = bm + wm + i * 16 + g * 8 + v;
        Y[row * D_ + col] = (_Float16)(acc[i][j][v] + bval);
      }
    }
}

// ---- K/V tile stage for flash attention: K async-DMA (natural), V manual transpose ----
__device__ __forceinline__ void load_kv_tile(const _Float16* __restrict__ Kg,
                                             const _Float16* __restrict__ Vg,
                                             size_t rowbase, int kt,
                                             _Float16* KsBuf, _Float16* VsBuf, int tid) {
#if USE_ASYNC
  {
    int r = tid >> 3, c8 = (tid & 7) * 8;          // 32 rows x 8 granules(16B) = 256 lanes
    async_cp16(&KsBuf[r * 72 + c8], &Kg[rowbase + (size_t)(kt + r) * D_ + c8]);
  }
#else
#pragma unroll
  for (int i = 0; i < 8; ++i) {
    int l = tid + i * 256;
    int r = l >> 6, c = l & 63;
    KsBuf[r * 72 + c] = Kg[rowbase + (size_t)(kt + r) * D_ + c];
  }
#endif
#pragma unroll
  for (int i = 0; i < 8; ++i) {                    // V 32x64 -> transposed [dk][key]
    int l = tid + i * 256;
    int r = l >> 6, c = l & 63;
    VsBuf[c * 40 + r] = Vg[rowbase + (size_t)(kt + r) * D_ + c];
  }
}

// ---------------- Kernel 2: flash attention per (b,h), 128 q-rows/block ----------------
__global__ __launch_bounds__(256)
void flash_attn(const _Float16* __restrict__ Qg, const _Float16* __restrict__ Kg,
                const _Float16* __restrict__ Vg, const int* __restrict__ mask,
                _Float16* __restrict__ Ctx) {
  __shared__ _Float16 Qs[128 * 72];          // [q][dk]
  __shared__ _Float16 Ks[2][32 * 72];        // double-buffered [key][dk]
  __shared__ _Float16 Vs[2][64 * 40];        // double-buffered [dk][key] (transposed)
  __shared__ _Float16 Ps[8][16 * 36];        // per-wave P tile [q16][key32]
  const int tid = threadIdx.x;
  const int lane = tid & 31;
  const int wid = tid >> 5;
  const int b = blockIdx.y >> 4;
  const int h = blockIdx.y & 15;
  const int q0 = blockIdx.x * 128;
  const int qw = q0 + wid * 16;
  const size_t rowbase = (size_t)(b * S_) * D_ + h * DK_;
  const int g = lane >> 4, nlo = lane & 15;
  const float scale = 0.125f;                // 1/sqrt(64)

  // ---- prologue: Q tile (one-time) + K/V tile 0 ----
#if USE_ASYNC
#pragma unroll
  for (int i = 0; i < 4; ++i) {              // Q: 128x64 halfs = 1024 granules of 16B
    int gi = tid + i * 256;
    int r = gi >> 3, c8 = (gi & 7) * 8;
    async_cp16(&Qs[r * 72 + c8], &Qg[rowbase + (size_t)(q0 + r) * D_ + c8]);
  }
#else
#pragma unroll
  for (int i = 0; i < 32; ++i) {
    int l = tid + i * 256;
    int r = l >> 6, c = l & 63;
    Qs[r * 72 + c] = Qg[rowbase + (size_t)(q0 + r) * D_ + c];
  }
#endif
  load_kv_tile(Kg, Vg, rowbase, 0, Ks[0], Vs[0], tid);
  ASYNC_WAIT();
  __syncthreads();

  v16h aq0 = load_A_frag(&Qs[(wid * 16) * 72], 72);       // dk 0..31
  v16h aq1 = load_A_frag(&Qs[(wid * 16) * 72 + 32], 72);  // dk 32..63

  v8f vzero = {};
  v8f ctx[4];
#pragma unroll
  for (int j = 0; j < 4; ++j) ctx[j] = vzero;
  float rowmax[8], rowsum[8];
#pragma unroll
  for (int v = 0; v < 8; ++v) { rowmax[v] = -3.0e38f; rowsum[v] = 0.0f; }

  for (int kt = 0; kt < S_; kt += 32) {
    const int cur = (kt >> 5) & 1;
    // prefetch next tile into the other buffer (overlaps DMA with compute below)
    if (kt + 32 < S_)
      load_kv_tile(Kg, Vg, rowbase, kt + 32, Ks[cur ^ 1], Vs[cur ^ 1], tid);

    // scores: S = Q(16x64) * K^T(64x32) as two 16x16 tiles, k split 2x32
    v8f s0 = vzero, s1 = vzero;
    v16h bk0 = load_B_frag(&Ks[cur][0], 72);
    v16h bk1 = load_B_frag(&Ks[cur][32], 72);
    s0 = __builtin_amdgcn_wmma_f32_16x16x32_f16(false, aq0, false, bk0, (short)0, s0, false, false);
    s0 = __builtin_amdgcn_wmma_f32_16x16x32_f16(false, aq1, false, bk1, (short)0, s0, false, false);
    v16h bk2 = load_B_frag(&Ks[cur][16 * 72], 72);
    v16h bk3 = load_B_frag(&Ks[cur][16 * 72 + 32], 72);
    s1 = __builtin_amdgcn_wmma_f32_16x16x32_f16(false, aq0, false, bk2, (short)0, s1, false, false);
    s1 = __builtin_amdgcn_wmma_f32_16x16x32_f16(false, aq1, false, bk3, (short)0, s1, false, false);

    // online softmax; this lane holds rows qw+8g..qw+8g+7, cols nlo / 16+nlo
#pragma unroll
    for (int v = 0; v < 8; ++v) {
      int qg = qw + g * 8 + v;
      const int* mrow = mask + (size_t)(b * S_ + qg) * S_ + kt;
      float x0 = s0[v] * scale;
      float x1 = s1[v] * scale;
      if (mrow[nlo] == 0)      x0 = NEGV;
      if (mrow[16 + nlo] == 0) x1 = NEGV;
      float mx   = red16_max(fmaxf(x0, x1));
      float mnew = fmaxf(rowmax[v], mx);
      float alpha = __expf(rowmax[v] - mnew);
      rowmax[v] = mnew;
      float p0 = __expf(x0 - mnew);
      float p1 = __expf(x1 - mnew);
      rowsum[v] = rowsum[v] * alpha + red16_sum(p0 + p1);
#pragma unroll
      for (int j = 0; j < 4; ++j) ctx[j][v] *= alpha;
      int m = g * 8 + v;
      Ps[wid][m * 36 + nlo]      = (_Float16)p0;
      Ps[wid][m * 36 + 16 + nlo] = (_Float16)p1;
    }

    // ctx += P(16x32) * V(32x64); same-wave LDS write->read is in-order (DScnt)
    v16h ap = load_A_frag(&Ps[wid][0], 36);
#pragma unroll
    for (int j = 0; j < 4; ++j) {
      v16h bvf = load_B_frag(&Vs[cur][(j * 16) * 40], 40);
      ctx[j] = __builtin_amdgcn_wmma_f32_16x16x32_f16(false, ap, false, bvf, (short)0, ctx[j], false, false);
    }

    ASYNC_WAIT();        // this wave's prefetch DMA done before the visibility barrier
    __syncthreads();     // next tile fully staged; buffer 'cur' free to overwrite
  }

#pragma unroll
  for (int v = 0; v < 8; ++v) {
    float inv = 1.0f / rowsum[v];
    int qg = qw + g * 8 + v;
#pragma unroll
    for (int j = 0; j < 4; ++j)
      Ctx[rowbase + (size_t)qg * D_ + j * 16 + nlo] = (_Float16)(ctx[j][v] * inv);
  }
}

// ---------------- Kernel 3: output projection (f16 ctx -> f32 out) ----------------
__global__ __launch_bounds__(256)
void out_gemm(const _Float16* __restrict__ Xh, const float* __restrict__ W,
              const float* __restrict__ bias, float* __restrict__ Y) {
  __shared__ _Float16 As[128 * 40];
  __shared__ _Float16 Bs[128 * 40];
  const int tid  = threadIdx.x;
  const int lane = tid & 31;
  const int wid  = tid >> 5;
  const int wm = (wid >> 2) * 64;
  const int wn = (wid & 3) * 32;
  const int bn = blockIdx.x * 128;
  const int bm = blockIdx.y * 128;

  v8f vzero = {};
  v8f acc[4][2];
#pragma unroll
  for (int i = 0; i < 4; ++i)
#pragma unroll
    for (int j = 0; j < 2; ++j) acc[i][j] = vzero;

  for (int k0 = 0; k0 < D_; k0 += 32) {
    __syncthreads();
#if USE_ASYNC
#pragma unroll
    for (int i = 0; i < 2; ++i) {            // A tile 128x32 f16 = 512 granules of 16B
      int gi = tid + i * 256;
      int r = gi >> 2, c8 = (gi & 3) * 8;
      async_cp16(&As[r * 40 + c8], &Xh[(bm + r) * D_ + k0 + c8]);
    }
#else
#pragma unroll
    for (int i = 0; i < 16; ++i) {
      int l = tid + i * 256;
      int r = l >> 5, c = l & 31;
      As[r * 40 + c] = Xh[(bm + r) * D_ + (k0 + c)];
    }
#endif
#pragma unroll
    for (int i = 0; i < 16; ++i) {           // Wo tile f32->f16 transpose: manual
      int l = tid + i * 256;
      int n = l & 127, kk = l >> 7;
      Bs[n * 40 + kk] = (_Float16)W[(k0 + kk) * D_ + (bn + n)];
    }
    ASYNC_WAIT();
    __syncthreads();
    v16h af[4], bf[2];
#pragma unroll
    for (int i = 0; i < 4; ++i) af[i] = load_A_frag(&As[(wm + i * 16) * 40], 40);
#pragma unroll
    for (int j = 0; j < 2; ++j) bf[j] = load_B_frag(&Bs[(wn + j * 16) * 40], 40);
#pragma unroll
    for (int i = 0; i < 4; ++i)
#pragma unroll
      for (int j = 0; j < 2; ++j)
        acc[i][j] = __builtin_amdgcn_wmma_f32_16x16x32_f16(
            false, af[i], false, bf[j], (short)0, acc[i][j], false, false);
  }

  const int g = lane >> 4, nlo = lane & 15;
#pragma unroll
  for (int i = 0; i < 4; ++i)
#pragma unroll
    for (int j = 0; j < 2; ++j) {
      int col = bn + wn + j * 16 + nlo;
      float bval = bias[col];
#pragma unroll
      for (int v = 0; v < 8; ++v) {
        int row = bm + wm + i * 16 + g * 8 + v;
        Y[row * D_ + col] = acc[i][j][v] + bval;
      }
    }
}

extern "C" void kernel_launch(void* const* d_in, const int* in_sizes, int n_in,
                              void* d_out, int out_size, void* d_ws, size_t ws_size,
                              hipStream_t stream) {
  const float* query = (const float*)d_in[0];
  const float* key   = (const float*)d_in[1];
  const float* value = (const float*)d_in[2];
  const int*   mask  = (const int*)d_in[3];
  const float* Wq = (const float*)d_in[4];
  const float* bq = (const float*)d_in[5];
  const float* Wk = (const float*)d_in[6];
  const float* bk = (const float*)d_in[7];
  const float* Wv = (const float*)d_in[8];
  const float* bv = (const float*)d_in[9];
  const float* Wo = (const float*)d_in[10];
  const float* bo = (const float*)d_in[11];

  char* ws = (char*)d_ws;
  const size_t plane = (size_t)B_ * S_ * D_ * sizeof(_Float16);  // 16 MB
  _Float16* qh = (_Float16*)(ws);
  _Float16* kh = (_Float16*)(ws + plane);
  _Float16* vh = (_Float16*)(ws + 2 * plane);
  _Float16* ch = (_Float16*)(ws + 3 * plane);

  dim3 blk(256);
  proj_gemm_f32<<<dim3(D_ / 128, (B_ * S_) / 128, 3), blk, 0, stream>>>(
      query, key, value, Wq, Wk, Wv, bq, bk, bv, qh, kh, vh);
  flash_attn<<<dim3(S_ / 128, B_ * H_), blk, 0, stream>>>(qh, kh, vh, mask, ch);
  out_gemm<<<dim3(D_ / 128, (B_ * S_) / 128), blk, 0, stream>>>(ch, Wo, bo, (float*)d_out);
}
